// FeedForwardApproximateBNN_82214263980126
// MI455X (gfx1250) — matrix operand order repaired and checked by reading, not compiled, not run on hardware
//
#include <hip/hip_runtime.h>
#include <stdint.h>

typedef __attribute__((ext_vector_type(16))) __bf16 v16bf;
typedef __attribute__((ext_vector_type(8)))  float  v8f;

#define BM 128
#define BN 256
#define KT 32

__device__ __forceinline__ uint32_t pack2bf(float lo, float hi) {
    uint32_t a = __float_as_uint(lo);
    uint32_t b = __float_as_uint(hi);
    a = (a + 0x7FFFu + ((a >> 16) & 1u)) >> 16;
    b = (b + 0x7FFFu + ((b >> 16) & 1u)) >> 16;
    return (a & 0xFFFFu) | (b << 16);
}

__device__ __forceinline__ uint16_t f2bf(float x) {
    uint32_t u = __float_as_uint(x);
    u = (u + 0x7FFFu + ((u >> 16) & 1u)) >> 16;
    return (uint16_t)u;
}

// Branchless per-channel activation using hardware transcendentals (v_exp_f32).
__device__ __forceinline__ float activate(float x, int idx) {
    float relu = fmaxf(x, 0.0f);
    float enx  = __expf(-x);                   // e^-x
    float sig  = 1.0f / (1.0f + enx);          // sigmoid
    float en2x = enx * enx;                    // e^-2x
    float th   = 2.0f / (1.0f + en2x) - 1.0f;  // tanh
    float lrl  = fmaxf(x, 0.1f * x);           // leaky relu(0.1)
    const float alpha = 1.6732632423543772f;
    const float scale = 1.0507009873554805f;
    float selu = scale * (x > 0.0f ? x : alpha * (__expf(x) - 1.0f));
    float r = selu;
    r = (idx == 3) ? lrl  : r;
    r = (idx == 2) ? th   : r;
    r = (idx == 1) ? sig  : r;
    r = (idx == 0) ? relu : r;
    return r;
}

__global__ void __launch_bounds__(256)
f32_to_bf16_kernel(const float* __restrict__ in, uint32_t* __restrict__ out, int n2) {
    int i = blockIdx.x * blockDim.x + threadIdx.x;
    if (i < n2) {
        float2 v = ((const float2*)in)[i];
        out[i] = pack2bf(v.x, v.y);
    }
}

union FragAB { uint4 q[2]; v16bf v; };

// C[M,N] = act(A[M,K](bf16) * Wbf[N,K]^T(bf16) + bias); store bf16 or f32.
// Block tile 128x256, 8 waves, each wave 64x64 = 4x4 WMMA 16x16x32 tiles.
__global__ void __launch_bounds__(256) __attribute__((amdgpu_waves_per_eu(2)))
bnn_layer_gemm(const uint16_t* __restrict__ Abf,
               const uint16_t* __restrict__ Wbf,
               const float*    __restrict__ bias,
               const int*      __restrict__ tf,
               uint16_t*       __restrict__ outBf,
               float*          __restrict__ outF,
               int M, int N, int K)
{
    __shared__ uint4 ldsA[(BM * KT) / 8];   // 512  x 16B = 8KB
    __shared__ uint4 ldsB[(BN * KT) / 8];   // 1024 x 16B = 16KB

    const int tid   = threadIdx.x;
    const int lane  = tid & 31;
    const int wave  = tid >> 5;      // 0..7
    const int wm    = wave & 1;      // M half (64 rows)
    const int wn    = wave >> 1;     // N quarter (64 cols)
    const int lhalf = lane & 15;
    const int kq    = lane >> 4;     // K-group select per ISA layout

    const int blockM = blockIdx.y * BM;
    const int blockN = blockIdx.x * BN;

    v8f acc[4][4];
    #pragma unroll
    for (int mt = 0; mt < 4; ++mt)
        #pragma unroll
        for (int nt = 0; nt < 4; ++nt)
            #pragma unroll
            for (int r = 0; r < 8; ++r)
                acc[mt][nt][r] = 0.0f;

    for (int k0 = 0; k0 < K; k0 += KT) {
        __syncthreads();
        // ---- global -> LDS copies (short-lived transients, no cross-barrier regs) ----
        #pragma unroll
        for (int i = 0; i < 2; ++i) {
            int c = tid + i * 256;
            ldsA[c] = ((const uint4*)(Abf + (size_t)(blockM + (c >> 2)) * K + k0))[c & 3];
        }
        #pragma unroll
        for (int i = 0; i < 4; ++i) {
            int c = tid + i * 256;
            ldsB[c] = ((const uint4*)(Wbf + (size_t)(blockN + (c >> 2)) * K + k0))[c & 3];
        }
        __syncthreads();

        // A fragments (16x32 bf16 layout: M=lane&15, K halves by lane group)
        FragAB a[4];
        #pragma unroll
        for (int mt = 0; mt < 4; ++mt) {
            int row = wm * 64 + mt * 16 + lhalf;
            a[mt].q[0] = ldsA[row * 4 + kq];
            a[mt].q[1] = ldsA[row * 4 + kq + 2];
        }
        // B fragments (32x16 bf16 layout: N=lane&15, K halves by lane group)
        #pragma unroll
        for (int nt = 0; nt < 4; ++nt) {
            FragAB b;
            int row = wn * 64 + nt * 16 + lhalf;
            b.q[0] = ldsB[row * 4 + kq];
            b.q[1] = ldsB[row * 4 + kq + 2];
            #pragma unroll
            for (int mt = 0; mt < 4; ++mt)
                acc[mt][nt] = __builtin_amdgcn_wmma_f32_16x16x32_bf16(
                    false, a[mt].v, false, b.v,
                    (short)0, acc[mt][nt], false, false);
        }
    }

    // ---- epilogue: bias + branchless per-channel activation ----
    #pragma unroll
    for (int nt = 0; nt < 4; ++nt) {
        int n = blockN + wn * 64 + nt * 16 + lhalf;
        float bv = bias[n];
        int  idx = tf[n];
        #pragma unroll
        for (int mt = 0; mt < 4; ++mt) {
            int mbase = blockM + wm * 64 + mt * 16 + kq * 8;  // C/D: lanes16-31 -> M+8
            #pragma unroll
            for (int r = 0; r < 8; ++r) {
                float v = activate(acc[mt][nt][r] + bv, idx);
                size_t off = (size_t)(mbase + r) * N + n;
                if (outBf) outBf[off] = f2bf(v);
                else       outF[off]  = v;
            }
        }
    }
}

extern "C" void kernel_launch(void* const* d_in, const int* in_sizes, int n_in,
                              void* d_out, int out_size, void* d_ws, size_t ws_size,
                              hipStream_t stream) {
    (void)in_sizes; (void)n_in; (void)out_size; (void)ws_size;

    const float* input = (const float*)d_in[0];
    const float* W_in  = (const float*)d_in[1];
    const float* b_in  = (const float*)d_in[2];
    const float* W_h   = (const float*)d_in[3];
    const float* b_h   = (const float*)d_in[4];
    const float* W_out = (const float*)d_in[5];
    const float* b_out = (const float*)d_in[6];
    const int*   tf_in = (const int*)d_in[7];
    const int*   tf_h  = (const int*)d_in[8];
    const int*   tf_out= (const int*)d_in[9];

    const int Mrows = 32 * 256;  // 8192
    const int Din = 512, X = 2048, Z = 4, Dout = 256;

    // workspace layout (bytes)
    char* ws = (char*)d_ws;
    size_t off = 0;
    uint16_t* bufIn  = (uint16_t*)(ws + off); off += (size_t)Mrows * Din * 2;   //  8.4 MB
    uint16_t* bufA   = (uint16_t*)(ws + off); off += (size_t)Mrows * X * 2;     // 33.6 MB
    uint16_t* bufB   = (uint16_t*)(ws + off); off += (size_t)Mrows * X * 2;     // 33.6 MB
    uint16_t* wInBf  = (uint16_t*)(ws + off); off += (size_t)X * Din * 2;       //  2.1 MB
    uint16_t* wHBf   = (uint16_t*)(ws + off); off += (size_t)Z * X * X * 2;     // 33.6 MB
    uint16_t* wOutBf = (uint16_t*)(ws + off); off += (size_t)Dout * X * 2;      //  1.0 MB

    // one-time fp32 -> bf16 conversions (out of the GEMM hot loops)
    {
        int n2 = Mrows * Din / 2;
        f32_to_bf16_kernel<<<(n2 + 255) / 256, 256, 0, stream>>>(input, (uint32_t*)bufIn, n2);
        n2 = X * Din / 2;
        f32_to_bf16_kernel<<<(n2 + 255) / 256, 256, 0, stream>>>(W_in, (uint32_t*)wInBf, n2);
        n2 = Z * X * X / 2;
        f32_to_bf16_kernel<<<(n2 + 255) / 256, 256, 0, stream>>>(W_h, (uint32_t*)wHBf, n2);
        n2 = Dout * X / 2;
        f32_to_bf16_kernel<<<(n2 + 255) / 256, 256, 0, stream>>>(W_out, (uint32_t*)wOutBf, n2);
    }

    dim3 blk(256);
    // input layer: [8192,512] x [512,2048]
    bnn_layer_gemm<<<dim3(X / BN, Mrows / BM), blk, 0, stream>>>(
        bufIn, wInBf, b_in, tf_in, bufA, nullptr, Mrows, X, Din);

    // hidden layers: 4 x [8192,2048] x [2048,2048]
    uint16_t* cur = bufA;
    uint16_t* nxt = bufB;
    for (int i = 0; i < Z; ++i) {
        bnn_layer_gemm<<<dim3(X / BN, Mrows / BM), blk, 0, stream>>>(
            cur, wHBf + (size_t)i * X * X, b_h + (size_t)i * X, tf_h + (size_t)i * X,
            nxt, nullptr, Mrows, X, X);
        uint16_t* t = cur; cur = nxt; nxt = t;
    }

    // output layer: [8192,2048] x [2048,256] -> fp32 d_out
    bnn_layer_gemm<<<dim3(Dout / BN, Mrows / BM), blk, 0, stream>>>(
        cur, wOutBf, b_out, tf_out, nullptr, (float*)d_out, Mrows, Dout, X);
}
